// GCN_36215164240762
// MI455X (gfx1250) — compile-verified, matrix operand order checked by reading, not run on hardware
//
#include <hip/hip_runtime.h>
#include <hip/hip_bf16.h>
#include <math.h>

typedef __attribute__((ext_vector_type(16))) _Float16 v16h;
typedef __attribute__((ext_vector_type(8)))  _Float16 v8h;
typedef __attribute__((ext_vector_type(8)))  float    v8f;

#define F_IN  500
#define KP1   512   // K padded for layer 1
#define HDIM  128
#define CDIM  40
#define NPAD2 64    // output cols padded for layer 2
#define KP2   128

// ---------------- small helper kernels ----------------

__global__ void k_init_deg(float* deg, int n) {
    int i = blockIdx.x * blockDim.x + threadIdx.x;
    if (i < n) deg[i] = 1.0f;   // self-loop contributes 1 to degree
}

__global__ void k_deg_edges(const int* __restrict__ dst, float* deg, int e) {
    int i = blockIdx.x * blockDim.x + threadIdx.x;
    if (i < e) atomicAdd(&deg[dst[i]], 1.0f);
}

__global__ void k_dinv(float* degdinv, int n) {
    int i = blockIdx.x * blockDim.x + threadIdx.x;
    if (i < n) degdinv[i] = rsqrtf(fmaxf(degdinv[i], 1e-12f));
}

// W1 [500 x 128] f32 -> W1t [128 x 512] f16, N-major, K zero-padded
__global__ void k_w1t(const float* __restrict__ W1, _Float16* __restrict__ W1t) {
    int i = blockIdx.x * blockDim.x + threadIdx.x;   // 128*512
    int nn = i >> 9, k = i & (KP1 - 1);
    float v = (k < F_IN) ? W1[k * HDIM + nn] : 0.0f;
    W1t[i] = (_Float16)v;
}

// W2 [128 x 40] f32 -> W2t [64 x 128] f16, N-major, N zero-padded
__global__ void k_w2t(const float* __restrict__ W2, _Float16* __restrict__ W2t) {
    int i = blockIdx.x * blockDim.x + threadIdx.x;   // 64*128
    int nn = i >> 7, k = i & (KP2 - 1);
    float v = (nn < CDIM) ? W2[k * CDIM + nn] : 0.0f;
    W2t[i] = (_Float16)v;
}

// ---------------- GEMM1: h1 = x @ W1  (f16 WMMA, f32 accum) ----------------
// One block = 16 rows x 128 cols. 8 waves, each owns one 16x16 tile over K=512.
__global__ void __launch_bounds__(256) k_gemm1(const float* __restrict__ x,
                                               const _Float16* __restrict__ W1t,
                                               float* __restrict__ h1, int n) {
    __shared__ __align__(16) _Float16 atile[16 * KP1];   // 16KB
    const int tid = threadIdx.x;
    const int r0  = blockIdx.x * 16;

    // stage A strip (16 x 500 f32 -> 16 x 512 f16, zero-padded)
    for (int i = tid; i < 16 * KP1; i += 256) {
        int row = i >> 9, col = i & (KP1 - 1);
        int gr = r0 + row;
        float v = (col < F_IN && gr < n) ? x[(size_t)gr * F_IN + col] : 0.0f;
        atile[i] = (_Float16)v;
    }
    __syncthreads();

    const int wave = tid >> 5, lane = tid & 31;
    const int m    = lane & 15;      // row (A) / col (B,D) within tile
    const int kh   = lane >> 4;      // which K-half this lane holds
    const int ncol = wave * 16 + m;  // global output column

    v8f acc = {};
    const _Float16* arow = &atile[m * KP1 + kh * 8];
    const _Float16* brow = &W1t[(size_t)ncol * KP1 + kh * 8];

#pragma unroll 4
    for (int ks = 0; ks < KP1 / 32; ++ks) {
        v8h alo = *(const v8h*)(arow + ks * 32);
        v8h ahi = *(const v8h*)(arow + ks * 32 + 16);
        v8h blo = *(const v8h*)(brow + ks * 32);
        v8h bhi = *(const v8h*)(brow + ks * 32 + 16);
        v16h a = __builtin_shufflevector(alo, ahi, 0,1,2,3,4,5,6,7,8,9,10,11,12,13,14,15);
        v16h b = __builtin_shufflevector(blo, bhi, 0,1,2,3,4,5,6,7,8,9,10,11,12,13,14,15);
        acc = __builtin_amdgcn_wmma_f32_16x16x32_f16(false, a, false, b,
                                                     (short)0, acc, false, false);
    }

    // D layout: lane holds col (lane&15), VGPR v holds row v + 8*(lane>>4)
    float* dcol = &h1[(size_t)(r0 + kh * 8) * HDIM + ncol];
    if (r0 + 16 <= n) {
        // whole tile in-bounds (always true when n % 16 == 0): no exec juggling
#pragma unroll
        for (int v = 0; v < 8; ++v) dcol[(size_t)v * HDIM] = acc[v];
    } else {
#pragma unroll
        for (int v = 0; v < 8; ++v) {
            if (r0 + kh * 8 + v < n) dcol[(size_t)v * HDIM] = acc[v];
        }
    }
}

// ---------------- GEMM2: h2 = a1 @ W2  (cols padded to 64) ----------------
// One block = 32 rows x 64 cols. Wave w -> (row-tile w>>2, col-tile w&3).
__global__ void __launch_bounds__(256) k_gemm2(const float* __restrict__ a1,
                                               const _Float16* __restrict__ W2t,
                                               float* __restrict__ h2, int n) {
    __shared__ __align__(16) _Float16 atile[32 * KP2];   // 8KB
    const int tid = threadIdx.x;
    const int r0  = blockIdx.x * 32;

    for (int i = tid; i < 32 * KP2; i += 256) {
        int row = i >> 7, col = i & (KP2 - 1);
        int gr = r0 + row;
        float v = (gr < n) ? a1[(size_t)gr * KP2 + col] : 0.0f;
        atile[i] = (_Float16)v;
    }
    __syncthreads();

    const int wave = tid >> 5, lane = tid & 31;
    const int rt = wave >> 2, ct = wave & 3;
    const int m = lane & 15, kh = lane >> 4;
    const int ncol = ct * 16 + m;

    v8f acc = {};
    const _Float16* arow = &atile[(rt * 16 + m) * KP2 + kh * 8];
    const _Float16* brow = &W2t[(size_t)ncol * KP2 + kh * 8];

#pragma unroll
    for (int ks = 0; ks < KP2 / 32; ++ks) {
        v8h alo = *(const v8h*)(arow + ks * 32);
        v8h ahi = *(const v8h*)(arow + ks * 32 + 16);
        v8h blo = *(const v8h*)(brow + ks * 32);
        v8h bhi = *(const v8h*)(brow + ks * 32 + 16);
        v16h a = __builtin_shufflevector(alo, ahi, 0,1,2,3,4,5,6,7,8,9,10,11,12,13,14,15);
        v16h b = __builtin_shufflevector(blo, bhi, 0,1,2,3,4,5,6,7,8,9,10,11,12,13,14,15);
        acc = __builtin_amdgcn_wmma_f32_16x16x32_f16(false, a, false, b,
                                                     (short)0, acc, false, false);
    }

    if (ncol < CDIM) {
        const int rb = r0 + rt * 16 + kh * 8;
        float* dcol = &h2[(size_t)rb * CDIM + ncol];
        if (r0 + 32 <= n) {
#pragma unroll
            for (int v = 0; v < 8; ++v) dcol[(size_t)v * CDIM] = acc[v];
        } else {
#pragma unroll
            for (int v = 0; v < 8; ++v) {
                if (rb + v < n) dcol[(size_t)v * CDIM] = acc[v];
            }
        }
    }
}

// ---------------- edge aggregation (L2-resident gather + scatter-add) ------
// Layer 1: 32 lanes/edge, 4 floats each (float4 gather, 4 f32 atomics).
__global__ void k_agg1(const int* __restrict__ src, const int* __restrict__ dst,
                       const float* __restrict__ dinv, const float* __restrict__ h1,
                       float* __restrict__ a1, int e, int n) {
    unsigned t = blockIdx.x * 256u + threadIdx.x;
    unsigned eidx = t >> 5;
    if (eidx >= (unsigned)(e + n)) return;
    int c0 = (t & 31) * 4;
    int s, d; float nrm;
    if (eidx < (unsigned)e) {
        s = src[eidx]; d = dst[eidx];
        nrm = dinv[s] * dinv[d];
    } else {
        s = d = (int)(eidx - (unsigned)e);
        float di = dinv[s];
        nrm = di * di;
    }
    const float4 hv = *(const float4*)&h1[(size_t)s * HDIM + c0];
    float* out = &a1[(size_t)d * HDIM + c0];
    atomicAdd(out + 0, hv.x * nrm);
    atomicAdd(out + 1, hv.y * nrm);
    atomicAdd(out + 2, hv.z * nrm);
    atomicAdd(out + 3, hv.w * nrm);
}

// Layer 2: 8 lanes/edge, 5 floats each (C=40).
__global__ void k_agg2(const int* __restrict__ src, const int* __restrict__ dst,
                       const float* __restrict__ dinv, const float* __restrict__ h2,
                       float* __restrict__ a2, int e, int n) {
    unsigned t = blockIdx.x * 256u + threadIdx.x;
    unsigned eidx = t >> 3;
    if (eidx >= (unsigned)(e + n)) return;
    int c0 = (t & 7) * 5;
    int s, d; float nrm;
    if (eidx < (unsigned)e) {
        s = src[eidx]; d = dst[eidx];
        nrm = dinv[s] * dinv[d];
    } else {
        s = d = (int)(eidx - (unsigned)e);
        float di = dinv[s];
        nrm = di * di;
    }
    const float* hp = &h2[(size_t)s * CDIM + c0];
    float* out = &a2[(size_t)d * CDIM + c0];
#pragma unroll
    for (int j = 0; j < 5; ++j) atomicAdd(out + j, hp[j] * nrm);
}

__global__ void k_bias_relu1(float* a1, const float* __restrict__ b1, unsigned total) {
    unsigned i = blockIdx.x * 256u + threadIdx.x;
    if (i < total) {
        float v = a1[i] + b1[i & (HDIM - 1)];
        a1[i] = v > 0.0f ? v : 0.0f;
    }
}

// out = log_softmax(relu(a2 + b2)), one thread per node (40 classes).
__global__ void k_out(const float* __restrict__ a2, const float* __restrict__ b2,
                      float* __restrict__ out, int n) {
    int i = blockIdx.x * blockDim.x + threadIdx.x;
    if (i >= n) return;
    float v[CDIM];
    float m = -INFINITY;
#pragma unroll
    for (int c = 0; c < CDIM; ++c) {
        float t = a2[(size_t)i * CDIM + c] + b2[c];
        t = t > 0.0f ? t : 0.0f;
        v[c] = t;
        m = fmaxf(m, t);
    }
    float s = 0.0f;
#pragma unroll
    for (int c = 0; c < CDIM; ++c) s += __expf(v[c] - m);
    float l = __logf(s);
#pragma unroll
    for (int c = 0; c < CDIM; ++c) out[(size_t)i * CDIM + c] = v[c] - m - l;
}

// ---------------- driver ----------------

extern "C" void kernel_launch(void* const* d_in, const int* in_sizes, int n_in,
                              void* d_out, int out_size, void* d_ws, size_t ws_size,
                              hipStream_t stream) {
    (void)n_in; (void)out_size; (void)ws_size;
    const float* x  = (const float*)d_in[0];
    const int*   ei = (const int*)d_in[1];
    const float* W1 = (const float*)d_in[2];
    const float* b1 = (const float*)d_in[3];
    const float* W2 = (const float*)d_in[4];
    const float* b2 = (const float*)d_in[5];
    float* out = (float*)d_out;

    const int N = in_sizes[0] / F_IN;
    const int E = in_sizes[1] / 2;
    const int* src = ei;
    const int* dst = ei + E;

    char* ws = (char*)d_ws;
    auto al = [](size_t v) { return (v + 255) & ~(size_t)255; };
    size_t oDinv = 0;
    size_t oW1t  = al((size_t)N * 4);
    size_t oW2t  = oW1t + (size_t)HDIM * KP1 * 2;
    size_t oH1   = al(oW2t + (size_t)NPAD2 * KP2 * 2);
    size_t oA1   = oH1 + (size_t)N * HDIM * 4;
    // after agg1, h1 region is dead: alias h2 and a2 into it
    size_t oH2 = oH1;
    size_t oA2 = oH1 + (size_t)N * CDIM * 4;

    float*    deg  = (float*)(ws + oDinv);   // becomes dinv in-place
    _Float16* W1t  = (_Float16*)(ws + oW1t);
    _Float16* W2t  = (_Float16*)(ws + oW2t);
    float*    h1   = (float*)(ws + oH1);
    float*    a1   = (float*)(ws + oA1);
    float*    h2   = (float*)(ws + oH2);
    float*    a2   = (float*)(ws + oA2);

    const int B = 256;
    int gN   = (N + B - 1) / B;
    int gE   = (E + B - 1) / B;
    long long en = (long long)E + N;

    // norm precompute
    k_init_deg <<<gN, B, 0, stream>>>(deg, N);
    k_w1t      <<<(HDIM * KP1) / B, B, 0, stream>>>(W1, W1t);
    k_w2t      <<<(NPAD2 * KP2) / B, B, 0, stream>>>(W2, W2t);
    k_deg_edges<<<gE, B, 0, stream>>>(dst, deg, E);
    k_dinv     <<<gN, B, 0, stream>>>(deg, N);

    // layer 1
    hipMemsetAsync(a1, 0, (size_t)N * HDIM * 4, stream);
    k_gemm1<<<(N + 15) / 16, B, 0, stream>>>(x, W1t, h1, N);
    {
        long long thr = en * 32;
        k_agg1<<<(unsigned)((thr + B - 1) / B), B, 0, stream>>>(src, dst, deg, h1, a1, E, N);
    }
    k_bias_relu1<<<(unsigned)(((long long)N * HDIM + B - 1) / B), B, 0, stream>>>(
        a1, b1, (unsigned)(N * HDIM));

    // layer 2 (h2/a2 alias dead h1 region)
    k_gemm2<<<(N + 31) / 32, B, 0, stream>>>(a1, W2t, h2, N);
    hipMemsetAsync(a2, 0, (size_t)N * CDIM * 4, stream);
    {
        long long thr = en * 8;
        k_agg2<<<(unsigned)((thr + B - 1) / B), B, 0, stream>>>(src, dst, deg, h2, a2, E, N);
    }
    k_out<<<gN, B, 0, stream>>>(a2, b2, out, N);
}
